// Encoder_odernn_15625091022981
// MI455X (gfx1250) — matrix operand order repaired and checked by reading, not compiled, not run on hardware
//
#include <hip/hip_runtime.h>
#include <cstddef>

// ---------------- problem constants ----------------
#define N_TRAJ 4096
#define T_STEPS 128
#define NDATA 64
#define INPW 128           // 2*NDATA
#define LAT 64
#define DCAT 256           // 2*LAT + INPW
#define NH 112             // NGRU/NODE/TZ = 100 padded to 112 (7 WMMA N-tiles)
#define NHP 128            // K padding for second GEMMs (4 WMMA K-tiles)
#define MTILE 16           // trajectories per block (one WMMA M-tile)
#define NTHREADS 256       // 8 wave32 waves

typedef __attribute__((ext_vector_type(16))) __bf16 v16bf;
typedef __attribute__((ext_vector_type(8)))  __bf16 v8bf;
typedef __attribute__((ext_vector_type(4)))  __bf16 v4bf;
typedef __attribute__((ext_vector_type(8)))  float  v8f;

// ---------------- LDS layout (bytes, all 32B-aligned) ----------------
constexpr size_t SZ_W1   = (size_t)DCAT * NH * 2;   // 57344
constexpr size_t SZ_ODW1 = (size_t)LAT  * NH * 2;   // 14336
constexpr size_t SZ_W2S  = (size_t)NHP  * LAT * 2;  // 16384
constexpr size_t SZ_NSW2 = (size_t)NHP  * NHP * 2;  // 32768

constexpr size_t OFS_UGW1 = 0;
constexpr size_t OFS_RGW1 = OFS_UGW1 + SZ_W1;
constexpr size_t OFS_NSW1 = OFS_RGW1 + SZ_W1;
constexpr size_t OFS_ODW1 = OFS_NSW1 + SZ_W1;
constexpr size_t OFS_UGW2 = OFS_ODW1 + SZ_ODW1;
constexpr size_t OFS_RGW2 = OFS_UGW2 + SZ_W2S;
constexpr size_t OFS_ODW2 = OFS_RGW2 + SZ_W2S;
constexpr size_t OFS_NSW2 = OFS_ODW2 + SZ_W2S;
constexpr size_t OFS_YC   = OFS_NSW2 + SZ_NSW2;                 // 16*256 bf16 = 8192
constexpr size_t OFS_HU   = OFS_YC   + (size_t)MTILE*DCAT*2;    // 16*128 bf16 = 4096
constexpr size_t OFS_HR   = OFS_HU   + (size_t)MTILE*NHP*2;
constexpr size_t OFS_Y    = OFS_HR   + (size_t)MTILE*NHP*2;     // f32 16*64
constexpr size_t OFS_S    = OFS_Y    + (size_t)MTILE*LAT*4;
constexpr size_t OFS_YODE = OFS_S    + (size_t)MTILE*LAT*4;
constexpr size_t OFS_U    = OFS_YODE + (size_t)MTILE*LAT*4;
constexpr size_t OFS_R    = OFS_U    + (size_t)MTILE*LAT*4;
constexpr size_t OFS_NSO  = OFS_R    + (size_t)MTILE*LAT*4;     // f32 16*128 = 8192
constexpr size_t OFS_UGB1 = OFS_NSO  + (size_t)MTILE*NHP*4;     // 112 f32
constexpr size_t OFS_RGB1 = OFS_UGB1 + NH*4;
constexpr size_t OFS_NSB1 = OFS_RGB1 + NH*4;
constexpr size_t OFS_ODB1 = OFS_NSB1 + NH*4;
constexpr size_t OFS_UGB2 = OFS_ODB1 + NH*4;                    // 64 f32
constexpr size_t OFS_RGB2 = OFS_UGB2 + LAT*4;
constexpr size_t OFS_ODB2 = OFS_RGB2 + LAT*4;
constexpr size_t OFS_NSB2 = OFS_ODB2 + LAT*4;                   // 128 f32
constexpr size_t SMEM_TOTAL = OFS_NSB2 + NHP*4;                 // 316416 B < 320 KB

// ---------------- WMMA fragment helpers (bf16 16x16x32) ----------------
// A: 16(M) x 32(K), ISA 16-bit A layout: lanes 0-15 rows M=0..15 with
// K = {0..7,16..23}; lanes 16-31 same rows with K = {8..15,24..31}.
// Two 16B contiguous runs per lane -> 2x ds_load_b128.
__device__ __forceinline__ v16bf load_a_frag(const __bf16* A, int lda, int lane) {
  const int row = lane & 15;
  const int kb  = (lane & 16) ? 8 : 0;
  const __bf16* p = A + row * lda + kb;
  const v8bf lo = *(const v8bf*)p;         // K = kb+0..7
  const v8bf hi = *(const v8bf*)(p + 16);  // K = kb+16..23
  return __builtin_shufflevector(lo, hi, 0, 1, 2, 3, 4, 5, 6, 7,
                                 8, 9, 10, 11, 12, 13, 14, 15);
}

// B fragments come from LDS pre-swizzled fragment-major: for fragment
// (kchunk kc, ntile nt), lane l holds 16 contiguous bf16 at
// ((kc*ntl + nt)*32 + l)*16  -> 2x ds_load_b128, zero shuffles at use.
__device__ __forceinline__ v16bf load_b_sw(const __bf16* Wsw, int ntl, int nt,
                                           int kc, int lane) {
  const __bf16* p = Wsw + ((((size_t)kc * ntl + nt) * 32 + lane) << 4);
  const v8bf lo = *(const v8bf*)p;
  const v8bf hi = *(const v8bf*)(p + 8);
  return __builtin_shufflevector(lo, hi, 0, 1, 2, 3, 4, 5, 6, 7,
                                 8, 9, 10, 11, 12, 13, 14, 15);
}

__device__ __forceinline__ v8f gemm_sw(const __bf16* A, int lda,
                                       const __bf16* Wsw, int ntl, int nt,
                                       int Kc, int lane, v8f acc) {
  for (int kc = 0; kc < Kc; ++kc) {
    const v16bf af = load_a_frag(A + kc * 32, lda, lane);
    const v16bf bf = load_b_sw(Wsw, ntl, nt, kc, lane);
    acc = __builtin_amdgcn_wmma_f32_16x16x32_bf16(false, af, false, bf,
                                                  (short)0, acc, false, false);
  }
  return acc;
}

// ---------------- weight staging (f32 -> bf16, swizzled, zero-padded) ----------------
// dst layout: fragment-major as consumed by load_b_sw. B-matrix ISA layout:
// lanes 0-15 col N=lane, K={0..7,16..23}; lanes 16-31 col N=lane-16, K={8..15,24..31}.
__device__ __forceinline__ void stage_w_sw(__bf16* dst, const float* src,
                                           int K, int N, int srcK, int srcN,
                                           int tid) {
  const int ntl = N >> 4;
  const int total = K * N;
  for (int idx = tid; idx < total; idx += NTHREADS) {
    const int f    = idx >> 9;          // 512 elems per 32x16 fragment
    const int r    = idx & 511;
    const int lane = r >> 4;
    const int e    = r & 15;
    const int kc   = f / ntl;
    const int nt   = f - kc * ntl;
    const int k    = kc * 32 + ((e & 8) ? 16 : 0) + ((lane & 16) ? 8 : 0) + (e & 7);
    const int n    = nt * 16 + (lane & 15);
    const float v  = (k < srcK && n < srcN) ? src[(size_t)k * srcN + n] : 0.0f;
    dst[idx] = (__bf16)v;
  }
}
__device__ __forceinline__ void load_bias_f32(float* dst, const float* src,
                                              int N, int srcN, int tid) {
  for (int i = tid; i < N; i += NTHREADS) dst[i] = (i < srcN) ? src[i] : 0.0f;
}

// branchless activations using raw v_rcp_f32 (exact at saturation limits:
// rcp(+inf)=0 -> tanh->1, exp(-inf)=0 -> tanh->-1 / sigmoid->{0,1}).
__device__ __forceinline__ float tanh_fast(float x) {
  return 1.0f - 2.0f * __builtin_amdgcn_rcpf(__expf(2.0f * x) + 1.0f);
}
__device__ __forceinline__ float sigmoid_fast(float x) {
  return __builtin_amdgcn_rcpf(1.0f + __expf(-x));
}
__device__ __forceinline__ v8f bcast8(float b) {
  v8f a = {b, b, b, b, b, b, b, b};
  return a;
}

// ---------------- persistent ODE-RNN kernel ----------------
__global__ void __launch_bounds__(NTHREADS, 1)
odernn_encoder(const float* __restrict__ data, const float* __restrict__ tps,
               const float* ug_w1, const float* ug_b1, const float* ug_w2, const float* ug_b2,
               const float* rg_w1, const float* rg_b1, const float* rg_w2, const float* rg_b2,
               const float* ns_w1, const float* ns_b1, const float* ns_w2, const float* ns_b2,
               const float* ode_w1, const float* ode_b1, const float* ode_w2, const float* ode_b2,
               const float* tz_w1, const float* tz_b1, const float* tz_w2, const float* tz_b2,
               float* __restrict__ out) {
  extern __shared__ __align__(32) char smem[];
  __bf16* ugw1_s = (__bf16*)(smem + OFS_UGW1);
  __bf16* rgw1_s = (__bf16*)(smem + OFS_RGW1);
  __bf16* nsw1_s = (__bf16*)(smem + OFS_NSW1);
  __bf16* odw1_s = (__bf16*)(smem + OFS_ODW1);
  __bf16* ugw2_s = (__bf16*)(smem + OFS_UGW2);
  __bf16* rgw2_s = (__bf16*)(smem + OFS_RGW2);
  __bf16* odw2_s = (__bf16*)(smem + OFS_ODW2);
  __bf16* nsw2_s = (__bf16*)(smem + OFS_NSW2);
  __bf16* yc     = (__bf16*)(smem + OFS_YC);   // [16][256] = [y|s|x] bf16
  __bf16* h_u    = (__bf16*)(smem + OFS_HU);   // [16][128] hidden (pad cols 0)
  __bf16* h_r    = (__bf16*)(smem + OFS_HR);
  float*  y_s    = (float*)(smem + OFS_Y);     // [16][64]
  float*  s_s    = (float*)(smem + OFS_S);
  float*  yode_s = (float*)(smem + OFS_YODE);
  float*  u_s    = (float*)(smem + OFS_U);
  float*  r_s    = (float*)(smem + OFS_R);
  float*  nso_s  = (float*)(smem + OFS_NSO);   // [16][128]
  float*  ugb1_s = (float*)(smem + OFS_UGB1);
  float*  rgb1_s = (float*)(smem + OFS_RGB1);
  float*  nsb1_s = (float*)(smem + OFS_NSB1);
  float*  odb1_s = (float*)(smem + OFS_ODB1);
  float*  ugb2_s = (float*)(smem + OFS_UGB2);
  float*  rgb2_s = (float*)(smem + OFS_RGB2);
  float*  odb2_s = (float*)(smem + OFS_ODB2);
  float*  nsb2_s = (float*)(smem + OFS_NSB2);

  const int tid  = threadIdx.x;
  const int lane = tid & 31;
  const int wave = tid >> 5;
  const int m0   = blockIdx.x * MTILE;
  const int col  = lane & 15;
  const int rb   = (lane & 16) ? 8 : 0;

  // ---- stage weights (one-time; swizzled into WMMA-B fragment layout) ----
  stage_w_sw(ugw1_s, ug_w1, DCAT, NH, DCAT, 100, tid);
  stage_w_sw(rgw1_s, rg_w1, DCAT, NH, DCAT, 100, tid);
  stage_w_sw(nsw1_s, ns_w1, DCAT, NH, DCAT, 100, tid);
  stage_w_sw(odw1_s, ode_w1, LAT, NH, LAT, 100, tid);
  stage_w_sw(ugw2_s, ug_w2, NHP, LAT, 100, LAT, tid);
  stage_w_sw(rgw2_s, rg_w2, NHP, LAT, 100, LAT, tid);
  stage_w_sw(odw2_s, ode_w2, NHP, LAT, 100, LAT, tid);
  stage_w_sw(nsw2_s, ns_w2, NHP, NHP, 100, 2 * LAT, tid);
  load_bias_f32(ugb1_s, ug_b1, NH, 100, tid);
  load_bias_f32(rgb1_s, rg_b1, NH, 100, tid);
  load_bias_f32(nsb1_s, ns_b1, NH, 100, tid);
  load_bias_f32(odb1_s, ode_b1, NH, 100, tid);
  load_bias_f32(ugb2_s, ug_b2, LAT, LAT, tid);
  load_bias_f32(rgb2_s, rg_b2, LAT, LAT, tid);
  load_bias_f32(odb2_s, ode_b2, LAT, LAT, tid);
  load_bias_f32(nsb2_s, ns_b2, NHP, NHP, tid);
  // zero state + hidden pads
  for (int i = tid; i < MTILE * NHP; i += NTHREADS) {
    h_u[i] = (__bf16)0.0f;
    h_r[i] = (__bf16)0.0f;
  }
  for (int i = tid; i < MTILE * LAT; i += NTHREADS) { y_s[i] = 0.0f; s_s[i] = 0.0f; }
  for (int i = tid; i < MTILE * 2 * LAT; i += NTHREADS) {
    const int row = i >> 7, c = i & 127;
    yc[row * DCAT + c] = (__bf16)0.0f;
  }
  __syncthreads();

  // ================= reverse-time scan =================
  for (int step = 0; step < T_STEPS; ++step) {
    const int tr = T_STEPS - 1 - step;
    const float dtv = (step == 0) ? -0.01f : (tps[tr] - tps[tr + 1]);

    // --- phase A: x_t -> yc[:,128:256] (bf16, float4 loads), prefetch next ---
    for (int e = tid; e < MTILE * INPW / 4; e += NTHREADS) {
      const int row = e >> 5, c4 = (e & 31) << 2;
      const float4 d =
          *(const float4*)&data[((size_t)(m0 + row) * T_STEPS + tr) * INPW + c4];
      v4bf o;
      o[0] = (__bf16)d.x; o[1] = (__bf16)d.y; o[2] = (__bf16)d.z; o[3] = (__bf16)d.w;
      *(v4bf*)&yc[row * DCAT + 128 + c4] = o;
    }
    if (tr > 0) {
      const int e = tid * 8;
      const int row = e >> 7, c = e & 127;
      __builtin_prefetch(&data[((size_t)(m0 + row) * T_STEPS + (tr - 1)) * INPW + c], 0, 0);
    }
    __syncthreads();

    // --- phase B: ODE hidden: h_u = tanh(y @ ode_w1 + b1) ---
    if (wave < 7) {
      const int n0 = wave * 16;
      v8f acc = gemm_sw(yc, DCAT, odw1_s, 7, wave, 2, lane, bcast8(odb1_s[n0 + col]));
#pragma unroll
      for (int rr = 0; rr < 8; ++rr)
        h_u[(rb + rr) * NHP + n0 + col] = (__bf16)tanh_fast(acc[rr]);
    }
    __syncthreads();

    // --- phase C: y_ode = y + (h_u @ ode_w2 + b2)*dt ; yc[:,0:64] = y_ode ---
    if (wave < 4) {
      const int n0 = wave * 16;
      v8f acc = gemm_sw(h_u, NHP, odw2_s, 4, wave, 4, lane, bcast8(odb2_s[n0 + col]));
#pragma unroll
      for (int rr = 0; rr < 8; ++rr) {
        const int row = rb + rr, c2 = n0 + col;
        const float v = y_s[row * LAT + c2] + acc[rr] * dtv;
        yode_s[row * LAT + c2] = v;
        yc[row * DCAT + c2] = (__bf16)v;   // cols 64..127 still hold bf16(s)
      }
    }
    __syncthreads();

    // --- phase D: gate hiddens (fused ug1|rg1): 14 N-tiles over 8 waves ---
    for (int t = wave; t < 14; t += 8) {
      const bool isU = (t < 7);
      const __bf16* W = isU ? ugw1_s : rgw1_s;
      const float*  B = isU ? ugb1_s : rgb1_s;
      __bf16*       H = isU ? h_u : h_r;
      const int nt = isU ? t : t - 7;
      const int n0 = nt * 16;
      v8f acc = gemm_sw(yc, DCAT, W, 7, nt, 8, lane, bcast8(B[n0 + col]));
#pragma unroll
      for (int rr = 0; rr < 8; ++rr)
        H[(rb + rr) * NHP + n0 + col] = (__bf16)tanh_fast(acc[rr]);
    }
    __syncthreads();

    // --- phase E: u = sigmoid(h_u @ ug_w2 + b2), r = sigmoid(h_r @ rg_w2 + b2) ---
    {
      const bool isU = (wave < 4);
      const __bf16* H = isU ? h_u : h_r;
      const __bf16* W = isU ? ugw2_s : rgw2_s;
      const float*  B = isU ? ugb2_s : rgb2_s;
      float*        O = isU ? u_s : r_s;
      const int nt = wave & 3;
      const int n0 = nt * 16;
      v8f acc = gemm_sw(H, NHP, W, 4, nt, 4, lane, bcast8(B[n0 + col]));
#pragma unroll
      for (int rr = 0; rr < 8; ++rr)
        O[(rb + rr) * LAT + n0 + col] = sigmoid_fast(acc[rr]);
    }
    __syncthreads();

    // --- phase F: yc[:,0:128] = bf16([y_ode*r | s*r]) ---
    for (int e = tid; e < MTILE * 2 * LAT; e += NTHREADS) {
      const int row = e >> 7, c = e & 127;
      const float rv = r_s[row * LAT + (c & 63)];
      const float v = (c < 64) ? yode_s[row * LAT + c] * rv
                               : s_s[row * LAT + (c - 64)] * rv;
      yc[row * DCAT + c] = (__bf16)v;
    }
    __syncthreads();

    // --- phase G: ns hidden ---
    if (wave < 7) {
      const int n0 = wave * 16;
      v8f acc = gemm_sw(yc, DCAT, nsw1_s, 7, wave, 8, lane, bcast8(nsb1_s[n0 + col]));
#pragma unroll
      for (int rr = 0; rr < 8; ++rr)
        h_u[(rb + rr) * NHP + n0 + col] = (__bf16)tanh_fast(acc[rr]);
    }
    __syncthreads();

    // --- phase H: ns = h_u @ ns_w2 + b2  (N=128, 8 waves) ---
    {
      const int n0 = wave * 16;
      v8f acc = gemm_sw(h_u, NHP, nsw2_s, 8, wave, 4, lane, bcast8(nsb2_s[n0 + col]));
#pragma unroll
      for (int rr = 0; rr < 8; ++rr)
        nso_s[(rb + rr) * NHP + n0 + col] = acc[rr];
    }
    __syncthreads();

    // --- phase I: GRU state update ---
    for (int e = tid; e < MTILE * LAT; e += NTHREADS) {
      const int row = e >> 6, c = e & 63;
      const float uu = u_s[e];
      const float yo = yode_s[e];
      const float ss = s_s[e];
      const float st = nso_s[row * NHP + c];
      const float sd = fabsf(nso_s[row * NHP + 64 + c]);
      const float m  = (float)yc[row * DCAT + 192 + c];   // mask bits (exact in bf16)
      float ny  = (1.0f - uu) * st + uu * yo;
      float nys = (1.0f - uu) * sd + uu * ss;
      ny  = m * ny  + (1.0f - m) * yo;
      nys = m * nys + (1.0f - m) * ss;
      nys = fabsf(nys);
      y_s[e] = ny;
      s_s[e] = nys;
      yc[row * DCAT + c]       = (__bf16)ny;
      yc[row * DCAT + 64 + c]  = (__bf16)nys;
    }
    __syncthreads();
  }

  // ================= final head: z = tanh([y,s]@tz_w1+b)@tz_w2+b =================
  stage_w_sw(ugw1_s, tz_w1, NHP, NH, 2 * LAT, 100, tid);   // reuse ug_w1 buffer
  stage_w_sw(rgw1_s, tz_w2, NHP, NHP, 100, 2 * LAT, tid);  // reuse rg_w1 buffer
  load_bias_f32(ugb1_s, tz_b1, NH, 100, tid);
  load_bias_f32(nsb2_s, tz_b2, NHP, NHP, tid);
  __syncthreads();

  if (wave < 7) {   // tz hidden: yc[:,0:128] holds bf16([y|s]) from last update
    const int n0 = wave * 16;
    v8f acc = gemm_sw(yc, DCAT, ugw1_s, 7, wave, 4, lane, bcast8(ugb1_s[n0 + col]));
#pragma unroll
    for (int rr = 0; rr < 8; ++rr)
      h_u[(rb + rr) * NHP + n0 + col] = (__bf16)tanh_fast(acc[rr]);
  }
  __syncthreads();

  {
    const int n0 = wave * 16;
    v8f acc = gemm_sw(h_u, NHP, rgw1_s, 8, wave, 4, lane, bcast8(nsb2_s[n0 + col]));
#pragma unroll
    for (int rr = 0; rr < 8; ++rr) {
      const int row = rb + rr, c2 = n0 + col;
      const float v = acc[rr];
      const size_t traj = (size_t)(m0 + row);
      if (c2 < LAT)
        out[traj * LAT + c2] = v;                                  // z0_mu
      else
        out[(size_t)N_TRAJ * LAT + traj * LAT + (c2 - LAT)] = fabsf(v);  // z0_std
    }
  }
}

// ---------------- host entry ----------------
extern "C" void kernel_launch(void* const* d_in, const int* in_sizes, int n_in,
                              void* d_out, int out_size, void* d_ws, size_t ws_size,
                              hipStream_t stream) {
  (void)in_sizes; (void)n_in; (void)out_size; (void)d_ws; (void)ws_size;
  const float* p[22];
  for (int i = 0; i < 22; ++i) p[i] = (const float*)d_in[i];
  dim3 grid(N_TRAJ / MTILE), block(NTHREADS);
  odernn_encoder<<<grid, block, SMEM_TOTAL, stream>>>(
      p[0], p[1], p[2], p[3], p[4], p[5], p[6], p[7], p[8], p[9], p[10], p[11],
      p[12], p[13], p[14], p[15], p[16], p[17], p[18], p[19], p[20], p[21],
      (float*)d_out);
}
